// StandardAttention_6176162972023
// MI455X (gfx1250) — matrix operand order repaired and checked by reading, not compiled, not run on hardware
//
#include <hip/hip_runtime.h>

typedef __attribute__((ext_vector_type(16))) _Float16 v16h;
typedef __attribute__((ext_vector_type(8)))  _Float16 v8h;
typedef __attribute__((ext_vector_type(4)))  _Float16 v4h;
typedef __attribute__((ext_vector_type(2)))  _Float16 h2;
typedef __attribute__((ext_vector_type(8)))  float    v8f;

namespace {
constexpr int kB = 2, kH = 16, kS = 2048, kD = 64;
constexpr int ROWS_WG = 64;   // 4 waves x 16 query rows
constexpr int KT = 32;        // keys per iteration
constexpr int KST = 80;       // halfs per s_K row (64 d + pad), 160B rows
constexpr int VST = 48;       // halfs per s_Vt row (32 keys + pad), 96B rows
constexpr int PST = 40;       // halfs per s_P row (32 cols + pad), 80B rows
}

// Load 16 halfs as two 16B chunks (both pointers 16B aligned).
__device__ __forceinline__ v16h ld16(const _Float16* lo, const _Float16* hi) {
  v8h a = *(const v8h*)lo;
  v8h b = *(const v8h*)hi;
  v16h r;
#pragma unroll
  for (int i = 0; i < 8; ++i) { r[i] = a[i]; r[i + 8] = b[i]; }
  return r;
}

__global__ __launch_bounds__(128, 1)
void fa_causal_f16wmma(const float* __restrict__ Q, const float* __restrict__ K,
                       const float* __restrict__ V, float* __restrict__ O) {
  __shared__ __attribute__((aligned(16))) _Float16 s_K[KT * KST];   // [key][d]
  __shared__ __attribute__((aligned(16))) _Float16 s_Vt[kD * VST];  // [d][key]
  __shared__ __attribute__((aligned(16))) _Float16 s_P[4 * 16 * PST];

  const int tid    = threadIdx.x;
  const int w      = tid >> 5;
  const int lane   = tid & 31;
  const int laneN  = lane & 15;
  const int laneHi = lane >> 4;

  const int nQB = kS / ROWS_WG;
  const int bh  = blockIdx.x / nQB;
  const int qb  = (blockIdx.x % nQB) * ROWS_WG;
  const size_t bhOff = (size_t)bh * kS * kD;

  // ---- Q fragments (A-layout 16x32), pre-scaled by log2(e)/sqrt(D) so the
  //      whole softmax runs in the exp2 domain (v_exp_f32 native). ----
  const float qscale = 0.125f * 1.4426950408889634f;
  const int rowA = qb + w * 16 + laneN;
  const float* qrow = Q + bhOff + (size_t)rowA * kD;
  v16h aQ[2];
#pragma unroll
  for (int f = 0; f < 2; ++f)
#pragma unroll
    for (int h = 0; h < 16; ++h) {
      int d = f * 32 + laneHi * 16 + h;
      aQ[f][h] = (_Float16)(qrow[d] * qscale);
    }

  // all-ones B fragment: acc[4] accumulates P*1 = softmax denominator
  v16h bOnes;
#pragma unroll
  for (int h = 0; h < 16; ++h) bOnes[h] = (_Float16)1.0f;

  v8f acc[5] = {};            // 4 output d-tiles + 1 denominator tile
  float mst[8];
#pragma unroll
  for (int r = 0; r < 8; ++r) mst[r] = -1e30f;

  const int wRow0 = qb + w * 16;              // wave's first query row
  const int rowC  = wRow0 + laneHi * 8;       // C-layout row base (add r)
  const int kEnd  = qb + ROWS_WG;

  for (int kb = 0; kb < kEnd; kb += KT) {
    __syncthreads();  // previous block's LDS reads done
    {   // ---- cooperative stage of K (f16) and V (f16, transposed) ----
      // Work unit: (key-pair kp, 4-wide d chunk). 16 pairs x 16 chunks = 256
      // units over 128 threads. All global loads issued first (batched waits),
      // then converted and stored: K rows as b64, V as packed key-pair b32.
      const float* kg = K + bhOff + (size_t)kb * kD;
      const float* vg = V + bhOff + (size_t)kb * kD;
      float4 kreg[2][2], vreg[2][2];
      int kpA[2], dA[2];
#pragma unroll
      for (int it = 0; it < 2; ++it) {
        const int idx = it * 128 + tid;
        const int kp  = idx >> 4;
        const int d   = (idx & 15) << 2;
        kpA[it] = kp; dA[it] = d;
        kreg[it][0] = *(const float4*)(kg + (2 * kp)     * kD + d);
        kreg[it][1] = *(const float4*)(kg + (2 * kp + 1) * kD + d);
        vreg[it][0] = *(const float4*)(vg + (2 * kp)     * kD + d);
        vreg[it][1] = *(const float4*)(vg + (2 * kp + 1) * kD + d);
      }
      // prefetch next 8KB K/V block into cache while this block is consumed
      if (kb + KT < kEnd) {
        const int pofs = (tid & 63) * 32;  // 64 x 128B lines per array
        __builtin_prefetch(kg + KT * kD + pofs, 0, 1);
        __builtin_prefetch(vg + KT * kD + pofs, 0, 1);
      }
#pragma unroll
      for (int it = 0; it < 2; ++it) {
        const int kp = kpA[it], d = dA[it];
        *(v4h*)&s_K[(2 * kp) * KST + d] =
            (v4h){(_Float16)kreg[it][0].x, (_Float16)kreg[it][0].y,
                  (_Float16)kreg[it][0].z, (_Float16)kreg[it][0].w};
        *(v4h*)&s_K[(2 * kp + 1) * KST + d] =
            (v4h){(_Float16)kreg[it][1].x, (_Float16)kreg[it][1].y,
                  (_Float16)kreg[it][1].z, (_Float16)kreg[it][1].w};
        const float va[4] = {vreg[it][0].x, vreg[it][0].y, vreg[it][0].z, vreg[it][0].w};
        const float vb[4] = {vreg[it][1].x, vreg[it][1].y, vreg[it][1].z, vreg[it][1].w};
#pragma unroll
        for (int j = 0; j < 4; ++j)
          *(h2*)&s_Vt[(d + j) * VST + 2 * kp] =
              (h2){(_Float16)va[j], (_Float16)vb[j]};
      }
    }
    __syncthreads();

    if (kb <= wRow0 + 15) {  // wave-uniform: skip fully-masked blocks
      // ---- scores: two 16x16 tiles (keys kb..+15, kb+16..+31) ----
      v8f c0 = {}, c1 = {};
#pragma unroll
      for (int f = 0; f < 2; ++f) {
        const _Float16* k0 = &s_K[laneN        * KST + f * 32 + laneHi * 16];
        const _Float16* k1 = &s_K[(16 + laneN) * KST + f * 32 + laneHi * 16];
        v16h bK0 = ld16(k0, k0 + 8);
        v16h bK1 = ld16(k1, k1 + 8);
        c0 = __builtin_amdgcn_wmma_f32_16x16x32_f16(false, aQ[f], false, bK0,
                                                    (short)0, c0, false, false);
        c1 = __builtin_amdgcn_wmma_f32_16x16x32_f16(false, aQ[f], false, bK1,
                                                    (short)0, c1, false, false);
      }

      float s0v[8], s1v[8];
#pragma unroll
      for (int r = 0; r < 8; ++r) { s0v[r] = c0[r]; s1v[r] = c1[r]; }

      // causal mask only when block can cross the diagonal (wave-uniform)
      if (kb + KT - 1 > wRow0) {
        const int col0 = kb + laneN, col1 = kb + 16 + laneN;
#pragma unroll
        for (int r = 0; r < 8; ++r) {
          const int rowg = rowC + r;
          if (col0 > rowg) s0v[r] = -1e30f;
          if (col1 > rowg) s1v[r] = -1e30f;
        }
      }

      // ---- packed-f16 row-max across 16 lanes: 4 independent xor chains ----
      h2 mx[4];
#pragma unroll
      for (int i = 0; i < 4; ++i)
        mx[i] = (h2){(_Float16)fmaxf(s0v[2 * i],     s1v[2 * i]),
                     (_Float16)fmaxf(s0v[2 * i + 1], s1v[2 * i + 1])};
#define SWZ_STEP(IMM)                                                        \
  _Pragma("unroll")                                                          \
  for (int i = 0; i < 4; ++i) {                                              \
    int t_ = __builtin_amdgcn_ds_swizzle(__builtin_bit_cast(int, mx[i]), IMM); \
    mx[i] = __builtin_elementwise_max(mx[i], __builtin_bit_cast(h2, t_));    \
  }
      SWZ_STEP(0x041F)  // xor 1
      SWZ_STEP(0x081F)  // xor 2
      SWZ_STEP(0x101F)  // xor 4
      SWZ_STEP(0x201F)  // xor 8
#undef SWZ_STEP

      // ---- online update (exp2 domain), P -> LDS as f16 ----
      _Float16* pw = &s_P[(w * 16 + laneHi * 8) * PST];
#pragma unroll
      for (int r = 0; r < 8; ++r) {
        float mnew = fmaxf(mst[r], (float)mx[r >> 1][r & 1]);
        float sc = __builtin_amdgcn_exp2f(mst[r] - mnew);
        mst[r] = mnew;
        float p0 = __builtin_amdgcn_exp2f(s0v[r] - mnew);
        float p1 = __builtin_amdgcn_exp2f(s1v[r] - mnew);
        acc[0][r] *= sc; acc[1][r] *= sc; acc[2][r] *= sc;
        acc[3][r] *= sc; acc[4][r] *= sc;
        pw[r * PST + laneN]      = (_Float16)p0;
        pw[r * PST + 16 + laneN] = (_Float16)p1;
      }
      // intra-wave LDS RAW: C-layout stores -> A-layout loads
      asm volatile("s_wait_dscnt 0x0" ::: "memory");

      // ---- P as A fragment (16x32); PV WMMAs + denominator WMMA ----
      const _Float16* pr = &s_P[(w * 16 + laneN) * PST];
      v16h aP = ld16(pr + laneHi * 16, pr + laneHi * 16 + 8);
#pragma unroll
      for (int t = 0; t < 4; ++t) {
        const _Float16* vt = &s_Vt[(t * 16 + laneN) * VST + laneHi * 16];
        v16h bV = ld16(vt, vt + 8);
        acc[t] = __builtin_amdgcn_wmma_f32_16x16x32_f16(false, aP, false, bV,
                                                        (short)0, acc[t], false, false);
      }
      acc[4] = __builtin_amdgcn_wmma_f32_16x16x32_f16(false, aP, false, bOnes,
                                                      (short)0, acc[4], false, false);
    }
  }

  // ---- normalize by the WMMA-accumulated denominator and store O (f32) ----
#pragma unroll
  for (int r = 0; r < 8; ++r) {
    const float inv = __builtin_amdgcn_rcpf(acc[4][r]);
    const size_t ro = bhOff + (size_t)(rowC + r) * kD;
#pragma unroll
    for (int t = 0; t < 4; ++t)
      O[ro + t * 16 + laneN] = acc[t][r] * inv;
  }
}

extern "C" void kernel_launch(void* const* d_in, const int* in_sizes, int n_in,
                              void* d_out, int out_size, void* d_ws, size_t ws_size,
                              hipStream_t stream) {
  (void)in_sizes; (void)n_in; (void)out_size; (void)d_ws; (void)ws_size;
  const float* q = (const float*)d_in[0];
  const float* k = (const float*)d_in[1];
  const float* v = (const float*)d_in[2];
  float* o = (float*)d_out;
  dim3 grid(kB * kH * (kS / ROWS_WG));
  dim3 block(128);
  hipLaunchKernelGGL(fa_causal_f16wmma, grid, block, 0, stream, q, k, v, o);
}
